// RotaryPEMultiHeadSelfAttention_9929964388671
// MI455X (gfx1250) — matrix operand order repaired and checked by reading, not compiled, not run on hardware
//
#include <hip/hip_runtime.h>
#include <hip/hip_bf16.h>

// ---- CDNA5 WMMA vector types ----
typedef __attribute__((ext_vector_type(16))) __bf16 v16bf;
typedef __attribute__((ext_vector_type(8)))  __bf16 bf16x8;
typedef __attribute__((ext_vector_type(8)))  float  v8f;
// ---- TDM descriptor vector types ----
typedef __attribute__((ext_vector_type(4))) unsigned int u32x4;
typedef __attribute__((ext_vector_type(8))) int          i32x8;
typedef __attribute__((ext_vector_type(4))) int          i32x4;

#define DMODEL 1024
#define NHEAD  16
#define DK     64
#define LSEQ   2048
#define BATCH  2
#define NROW   (LSEQ*BATCH)   // 4096 rows of x (l-major, b-minor)
#define NE     (3*DMODEL)     // 3072 qkv columns

#if __has_include(<hip/amd_detail/amd_gfx1250_TDM.h>)
#define TDM_6ARG 1
#endif

// =====================================================================
// TDM helper: issue a 2-D tile load (bf16 elements) into LDS.
// D# layout per CDNA5 ISA 8.3-8.6 (group0 128b, group1 256b; groups 2/3
// zero => 2-D tensor).  Issued per-wave (EXEC ignored by TENSOR ops);
// completion tracked with TENSORcnt.
// =====================================================================
__device__ __forceinline__ void tdm_load_2d_bf16(unsigned lds_off,
                                                 const void* gaddr,
                                                 unsigned tensor_w,   // dim0 len (elems)
                                                 unsigned tensor_h,   // dim1 len (rows)
                                                 unsigned tile_w,     // tile dim0 (elems)
                                                 unsigned tile_h,     // tile dim1 (rows)
                                                 unsigned row_stride) // dim0 stride (elems)
{
  unsigned long long ga = (unsigned long long)gaddr;
  u32x4 g0;
  g0[0] = 1u;                                            // count=1, user D#
  g0[1] = lds_off;                                       // lds_addr (bytes)
  g0[2] = (unsigned)(ga & 0xFFFFFFFFu);                  // global_addr[31:0]
  g0[3] = (unsigned)((ga >> 32) & 0x01FFFFFFu)           // global_addr[56:32]
        | (2u << 30);                                    // type=2 ("image")
  i32x8 g1;
  g1[0] = (int)(1u << 16);                               // data_size=1 (2 bytes)
  g1[1] = (int)((tensor_w & 0xFFFFu) << 16);             // tensor_dim0[15:0]
  g1[2] = (int)(((tensor_w >> 16) & 0xFFFFu)
        |       ((tensor_h & 0xFFFFu) << 16));           // dim0 hi | dim1 lo
  g1[3] = (int)(((tensor_h >> 16) & 0xFFFFu)
        |       ((tile_w & 0xFFFFu) << 16));             // dim1 hi | tile_dim0
  g1[4] = (int)(tile_h & 0xFFFFu);                       // tile_dim1 (tile_dim2=0)
  g1[5] = (int)row_stride;                               // tensor_dim0_stride[31:0]
  g1[6] = 0;                                             // stride hi | dim1_stride lo
  g1[7] = 0;
  i32x4 z4 = {0, 0, 0, 0};
#ifdef TDM_6ARG
  i32x8 z8 = {0, 0, 0, 0, 0, 0, 0, 0};
  __builtin_amdgcn_tensor_load_to_lds(g0, g1, z4, z4, z8, 0);
#else
  __builtin_amdgcn_tensor_load_to_lds(g0, g1, z4, z4, 0);
#endif
}

__device__ __forceinline__ unsigned lds_offset_u32(const void* p) {
  // flat LDS address carries the wave-relative LDS byte offset in addr[31:0]
  return (unsigned)(unsigned long long)(uintptr_t)p;
}

// =====================================================================
// Kernel 1: fp32 -> bf16 conversion of x and in_proj_weight
// =====================================================================
__global__ __launch_bounds__(256) void cvt_kernel(const float* __restrict__ x,
                                                  const float* __restrict__ w,
                                                  __bf16* __restrict__ xb,
                                                  __bf16* __restrict__ wb,
                                                  int nx, int nw) {
  int i = blockIdx.x * 256 + threadIdx.x;
  if (i < nx) xb[i] = (__bf16)x[i];
  if (i < nw) wb[i] = (__bf16)w[i];
}

// =====================================================================
// Kernel 2: QKV GEMM  qkv[row][e] = sum_d xb[row][d] * wb[e][d]
// M=4096, N=3072, K=1024.  TDM double-buffered LDS pipeline:
//   wave0 issues tensor_load_to_lds for A(64x32) and B(128x32) tiles,
//   s_wait_tensorcnt + barrier, all 8 waves consume via ds_load_b128 and
//   V_WMMA_F32_16X16X32_BF16 (fp32 accumulate).
// Block = 8 waves (2 M x 4 N), wave = 2x2 16x16 tiles.
// =====================================================================
__global__ __launch_bounds__(256) void qkv_gemm(const __bf16* __restrict__ xb,
                                                const __bf16* __restrict__ wb,
                                                float* __restrict__ qkv) {
  __shared__ __bf16 ldsA[2][64 * 32];    // [buf][m][k]   4 KB each
  __shared__ __bf16 ldsB[2][128 * 32];   // [buf][n][k]   8 KB each

  const int lane = threadIdx.x & 31;
  const int wave = threadIdx.x >> 5;
  const int wm = wave & 1;        // 0..1  (M)
  const int wn = wave >> 1;       // 0..3  (N)
  const int m_blk = blockIdx.y * 64;
  const int n_blk = blockIdx.x * 128;
  const int l15 = lane & 15;
  const bool hi  = lane >= 16;

  v8f c00 = {0,0,0,0,0,0,0,0}, c01 = c00, c10 = c00, c11 = c00;

  // ---- prologue: stage K-chunk 0 into buffer 0 ----
  if (wave == 0) {
    tdm_load_2d_bf16(lds_offset_u32(&ldsA[0][0]),
                     (const char*)xb + 2u * ((size_t)m_blk * DMODEL + 0),
                     DMODEL, NROW, 32, 64, DMODEL);
    tdm_load_2d_bf16(lds_offset_u32(&ldsB[0][0]),
                     (const char*)wb + 2u * ((size_t)n_blk * DMODEL + 0),
                     DMODEL, NE, 32, 128, DMODEL);
  }

  int buf = 0;
  for (int k0 = 0; k0 < DMODEL; k0 += 32, buf ^= 1) {
    if (wave == 0) {
      if (k0 + 32 < DMODEL) {
        // stage next K-chunk into the other buffer, then wait for current
        // buffer only (TDM ops complete in order per wave)
        tdm_load_2d_bf16(lds_offset_u32(&ldsA[buf ^ 1][0]),
                         (const char*)xb + 2u * ((size_t)m_blk * DMODEL + k0 + 32),
                         DMODEL, NROW, 32, 64, DMODEL);
        tdm_load_2d_bf16(lds_offset_u32(&ldsB[buf ^ 1][0]),
                         (const char*)wb + 2u * ((size_t)n_blk * DMODEL + k0 + 32),
                         DMODEL, NE, 32, 128, DMODEL);
        __builtin_amdgcn_s_wait_tensorcnt(2);
      } else {
        __builtin_amdgcn_s_wait_tensorcnt(0);
      }
    }
    __syncthreads();   // current buffer visible to all waves

    const __bf16* A = &ldsA[buf][0];   // [64][32]
    const __bf16* B = &ldsB[buf][0];   // [128][32]

    // A fragments (16x32): lane<16 row m, k 0..15; lane>=16 same row,
    // k 16..31
    v16bf a0, a1, b0, b1;
    {
      union { v16bf v; bf16x8 h[2]; } u;
      const __bf16* p = A + (wm * 32 + l15) * 32 + (hi ? 16 : 0);
      u.h[0] = *(const bf16x8*)p;
      u.h[1] = *(const bf16x8*)(p + 8);
      a0 = u.v;
      p += 16 * 32;
      u.h[0] = *(const bf16x8*)p;
      u.h[1] = *(const bf16x8*)(p + 8);
      a1 = u.v;
    }
    // B fragments (32x16): lane<16 col n, k 0..15; lane>=16 col n-16, k 16..31
    {
      union { v16bf v; bf16x8 h[2]; } u;
      const __bf16* p = B + (wn * 32 + l15) * 32 + (hi ? 16 : 0);
      u.h[0] = *(const bf16x8*)p;
      u.h[1] = *(const bf16x8*)(p + 8);
      b0 = u.v;
      p += 16 * 32;
      u.h[0] = *(const bf16x8*)p;
      u.h[1] = *(const bf16x8*)(p + 8);
      b1 = u.v;
    }
    c00 = __builtin_amdgcn_wmma_f32_16x16x32_bf16(false, a0, false, b0, (short)0, c00, false, false);
    c01 = __builtin_amdgcn_wmma_f32_16x16x32_bf16(false, a0, false, b1, (short)0, c01, false, false);
    c10 = __builtin_amdgcn_wmma_f32_16x16x32_bf16(false, a1, false, b0, (short)0, c10, false, false);
    c11 = __builtin_amdgcn_wmma_f32_16x16x32_bf16(false, a1, false, b1, (short)0, c11, false, false);

    __syncthreads();   // all waves done reading before TDM overwrites buffers
  }

  // C/D layout: lane<16 -> rows 0..7 (VGPR r), N=lane; lane>=16 -> rows 8..15
  const int roff = hi ? 8 : 0;
  const int m_base = m_blk + wm * 32;
  const int n_base = n_blk + wn * 32;
#pragma unroll
  for (int r = 0; r < 8; ++r) {
    qkv[(size_t)(m_base +  0 + roff + r) * NE + n_base +  0 + l15] = c00[r];
    qkv[(size_t)(m_base +  0 + roff + r) * NE + n_base + 16 + l15] = c01[r];
    qkv[(size_t)(m_base + 16 + roff + r) * NE + n_base +  0 + l15] = c10[r];
    qkv[(size_t)(m_base + 16 + roff + r) * NE + n_base + 16 + l15] = c11[r];
  }
}

// =====================================================================
// Kernel 3: RoPE + layout change.
//   q,k: rotate pairs (d, d+16) of first 32 dims by angle l * 10000^(-t/16).
//   q is pre-scaled by 1/sqrt(64).
//   q,k -> bf16 [hb][L][64] row-major; v -> bf16 transposed [hb][64][L].
// =====================================================================
__global__ __launch_bounds__(256) void rope_kernel(const float* __restrict__ qkv,
                                                   __bf16* __restrict__ qb,
                                                   __bf16* __restrict__ kb,
                                                   __bf16* __restrict__ vtb) {
  int idx = blockIdx.x * 256 + threadIdx.x;   // 0 .. 16*2*2048*64-1
  int d = idx & 63;
  int l = (idx >> 6) & 2047;
  int b = (idx >> 17) & 1;
  int h = idx >> 18;
  size_t base = (size_t)(l * BATCH + b) * NE + h * (3 * DK);

  float qv = qkv[base + d];
  float kv = qkv[base + DK + d];
  float vv = qkv[base + 2 * DK + d];

  if (d < 32) {
    int t = d & 15;
    float theta = __powf(10000.f, -(float)t * (1.f / 16.f));
    float ang = (float)l * theta;
    float sn, cs;
    __sincosf(ang, &sn, &cs);
    float qp = qkv[base + (d ^ 16)];
    float kp = qkv[base + DK + (d ^ 16)];
    if (d < 16) { qv = qv * cs - qp * sn; kv = kv * cs - kp * sn; }
    else        { qv = qv * cs + qp * sn; kv = kv * cs + kp * sn; }
  }
  qv *= 0.125f;  // fold softmax scale 1/sqrt(DK) into q

  int hb = h * BATCH + b;
  size_t o = ((size_t)hb * LSEQ + l) * DK + d;
  qb[o] = (__bf16)qv;
  kb[o] = (__bf16)kv;
  vtb[((size_t)hb * DK + d) * LSEQ + l] = (__bf16)vv;
}

// =====================================================================
// Kernel 4: flash attention per (h,b).  One wave owns a 16-row q tile,
// streams K/V in 32-column blocks with online softmax.
//   S tile (16x32) : 4 WMMAs;  PV update : 4 WMMAs
// P goes D-layout -> A-layout via per-wave LDS staging + s_wait_dscnt.
// =====================================================================
__global__ __launch_bounds__(256) void attn_kernel(const __bf16* __restrict__ qb,
                                                   const __bf16* __restrict__ kb,
                                                   const __bf16* __restrict__ vtb,
                                                   float* __restrict__ out) {
  __shared__ __bf16 lds_p[8][16 * 32];   // 1 KB per wave P staging

  const int lane = threadIdx.x & 31;
  const int wave = threadIdx.x >> 5;
  const int l15  = lane & 15;
  const bool hi  = lane >= 16;

  const int hb = blockIdx.y;             // 0..31
  const int h  = hb >> 1, b = hb & 1;
  const int q0 = (blockIdx.x * 8 + wave) * 16;   // q-tile start row

  const __bf16* qbase = qb  + (size_t)hb * LSEQ * DK;
  const __bf16* kbase = kb  + (size_t)hb * LSEQ * DK;
  const __bf16* vbase = vtb + (size_t)hb * DK * LSEQ;

  // q A-fragments for d in [0,32) and [32,64)
  v16bf qf0, qf1;
  {
    union { v16bf v; bf16x8 hh[2]; } u;
    const __bf16* p = qbase + (size_t)(q0 + l15) * DK + (hi ? 16 : 0);
    u.hh[0] = *(const bf16x8*)p;
    u.hh[1] = *(const bf16x8*)(p + 8);
    qf0 = u.v;
    u.hh[0] = *(const bf16x8*)(p + 32);
    u.hh[1] = *(const bf16x8*)(p + 40);
    qf1 = u.v;
  }

  v8f acc0 = {0,0,0,0,0,0,0,0}, acc1 = acc0, acc2 = acc0, acc3 = acc0;
  float mrow[8], lrow[8];
#pragma unroll
  for (int r = 0; r < 8; ++r) { mrow[r] = -3.0e38f; lrow[r] = 0.f; }

  __bf16* lp = lds_p[wave];

  for (int kblk = 0; kblk < LSEQ; kblk += 32) {
    // prefetch next K/V block while computing this one
    if (kblk + 32 < LSEQ) {
      __builtin_prefetch(kbase + (size_t)(kblk + 32 + l15) * DK, 0, 3);
      __builtin_prefetch(vbase + (size_t)l15 * LSEQ + kblk + 32, 0, 3);
    }

    // ---- scores: S(16x32) = q(16x64) . K^T ----
    v8f s0 = {0,0,0,0,0,0,0,0}, s1 = s0;
    {
      union { v16bf v; bf16x8 hh[2]; } u;
      const __bf16* p = kbase + (size_t)(kblk + l15) * DK + (hi ? 16 : 0);
      u.hh[0] = *(const bf16x8*)p;        u.hh[1] = *(const bf16x8*)(p + 8);
      v16bf kd0 = u.v;                    // d 0..31
      u.hh[0] = *(const bf16x8*)(p + 32); u.hh[1] = *(const bf16x8*)(p + 40);
      v16bf kd1 = u.v;                    // d 32..63
      s0 = __builtin_amdgcn_wmma_f32_16x16x32_bf16(false, qf0, false, kd0, (short)0, s0, false, false);
      s0 = __builtin_amdgcn_wmma_f32_16x16x32_bf16(false, qf1, false, kd1, (short)0, s0, false, false);
      p += (size_t)16 * DK;
      u.hh[0] = *(const bf16x8*)p;        u.hh[1] = *(const bf16x8*)(p + 8);
      kd0 = u.v;
      u.hh[0] = *(const bf16x8*)(p + 32); u.hh[1] = *(const bf16x8*)(p + 40);
      kd1 = u.v;
      s1 = __builtin_amdgcn_wmma_f32_16x16x32_bf16(false, qf0, false, kd0, (short)0, s1, false, false);
      s1 = __builtin_amdgcn_wmma_f32_16x16x32_bf16(false, qf1, false, kd1, (short)0, s1, false, false);
    }

    // ---- online softmax over this 32-column chunk ----
    float t[8];
#pragma unroll
    for (int r = 0; r < 8; ++r) t[r] = fmaxf(s0[r], s1[r]);
#pragma unroll
    for (int m = 1; m < 16; m <<= 1)
#pragma unroll
      for (int r = 0; r < 8; ++r) t[r] = fmaxf(t[r], __shfl_xor(t[r], m, 32));

    float p0[8], p1[8], rs[8];
#pragma unroll
    for (int r = 0; r < 8; ++r) {
      float mn = fmaxf(mrow[r], t[r]);
      float alpha = __expf(mrow[r] - mn);
      p0[r] = __expf(s0[r] - mn);
      p1[r] = __expf(s1[r] - mn);
      rs[r] = p0[r] + p1[r];
      mrow[r] = mn;
      lrow[r] *= alpha;
      acc0[r] *= alpha; acc1[r] *= alpha; acc2[r] *= alpha; acc3[r] *= alpha;
    }
#pragma unroll
    for (int m = 1; m < 16; m <<= 1)
#pragma unroll
      for (int r = 0; r < 8; ++r) rs[r] += __shfl_xor(rs[r], m, 32);
#pragma unroll
    for (int r = 0; r < 8; ++r) lrow[r] += rs[r];

    // ---- stage P (bf16) to LDS in row-major [16][32], re-read as A frag ----
    {
      const int rb = hi ? 8 : 0;
#pragma unroll
      for (int r = 0; r < 8; ++r) {
        lp[(rb + r) * 32 + l15]      = (__bf16)p0[r];   // tile0 cols 0..15
        lp[(rb + r) * 32 + 16 + l15] = (__bf16)p1[r];   // tile1 cols 16..31
      }
    }
    asm volatile("s_wait_dscnt 0" ::: "memory");   // cross-lane LDS visibility (wave-local)
    v16bf pf;
    {
      union { v16bf v; bf16x8 hh[2]; } u;
      const __bf16* p = lp + l15 * 32 + (hi ? 16 : 0);
      u.hh[0] = *(const bf16x8*)p;
      u.hh[1] = *(const bf16x8*)(p + 8);
      pf = u.v;
    }

    // ---- PV: acc(16x64) += P(16x32) . V(32x64)  (V stored transposed) ----
    {
      union { v16bf v; bf16x8 hh[2]; } u;
      const __bf16* p = vbase + (size_t)l15 * LSEQ + kblk + (hi ? 16 : 0);
      u.hh[0] = *(const bf16x8*)p;  u.hh[1] = *(const bf16x8*)(p + 8);
      acc0 = __builtin_amdgcn_wmma_f32_16x16x32_bf16(false, pf, false, u.v, (short)0, acc0, false, false);
      p += (size_t)16 * LSEQ;
      u.hh[0] = *(const bf16x8*)p;  u.hh[1] = *(const bf16x8*)(p + 8);
      acc1 = __builtin_amdgcn_wmma_f32_16x16x32_bf16(false, pf, false, u.v, (short)0, acc1, false, false);
      p += (size_t)16 * LSEQ;
      u.hh[0] = *(const bf16x8*)p;  u.hh[1] = *(const bf16x8*)(p + 8);
      acc2 = __builtin_amdgcn_wmma_f32_16x16x32_bf16(false, pf, false, u.v, (short)0, acc2, false, false);
      p += (size_t)16 * LSEQ;
      u.hh[0] = *(const bf16x8*)p;  u.hh[1] = *(const bf16x8*)(p + 8);
      acc3 = __builtin_amdgcn_wmma_f32_16x16x32_bf16(false, pf, false, u.v, (short)0, acc3, false, false);
    }
  }

  // ---- epilogue: out[l][b][h*64+d] = acc / l ----
  const int rb = hi ? 8 : 0;
#pragma unroll
  for (int r = 0; r < 8; ++r) {
    float inv = 1.f / lrow[r];
    size_t row = (size_t)(q0 + rb + r);
    size_t o = row * (BATCH * DMODEL) + (size_t)b * DMODEL + h * DK + l15;
    out[o +  0] = acc0[r] * inv;
    out[o + 16] = acc1[r] * inv;
    out[o + 32] = acc2[r] * inv;
    out[o + 48] = acc3[r] * inv;
  }
}

// =====================================================================
// Host launcher
// =====================================================================
extern "C" void kernel_launch(void* const* d_in, const int* in_sizes, int n_in,
                              void* d_out, int out_size, void* d_ws, size_t ws_size,
                              hipStream_t stream) {
  const float* x = (const float*)d_in[0];          // [2048,2,1024]
  const float* w = (const float*)d_in[1];          // [3072,1024]
  float* out = (float*)d_out;                      // [2048,2,1024]

  char* ws = (char*)d_ws;
  size_t off = 0;
  __bf16* xb  = (__bf16*)(ws + off); off += (size_t)NROW * DMODEL * 2;   // 8.4 MB
  __bf16* wb  = (__bf16*)(ws + off); off += (size_t)NE   * DMODEL * 2;   // 6.3 MB
  float*  qkv = (float*) (ws + off); off += (size_t)NROW * NE * 4;       // 50.3 MB
  __bf16* qb  = (__bf16*)(ws + off); off += (size_t)NHEAD * BATCH * LSEQ * DK * 2;
  __bf16* kb  = (__bf16*)(ws + off); off += (size_t)NHEAD * BATCH * LSEQ * DK * 2;
  __bf16* vtb = (__bf16*)(ws + off); off += (size_t)NHEAD * BATCH * LSEQ * DK * 2;

  // 1) convert to bf16
  cvt_kernel<<<(NROW * DMODEL) / 256, 256, 0, stream>>>(x, w, xb, wb,
                                                        NROW * DMODEL, NE * DMODEL);
  // 2) QKV GEMM: grid (N/128, M/64), TDM-fed LDS pipeline
  qkv_gemm<<<dim3(NE / 128, NROW / 64), 256, 0, stream>>>(xb, wb, qkv);
  // 3) RoPE + relayout: one thread per q/k/v element
  rope_kernel<<<(NHEAD * BATCH * LSEQ * DK) / 256, 256, 0, stream>>>(qkv, qb, kb, vtb);
  // 4) attention: grid (128 q-tiles / 8 waves, 32 head-batch pairs)
  attn_kernel<<<dim3(LSEQ / 16 / 8, NHEAD * BATCH), 256, 0, stream>>>(qb, kb, vtb, out);
}
